// LSTM_quantized_cell_80058190397494
// MI455X (gfx1250) — compile-verified, matrix operand order checked by reading, not run on hardware
//
#include <hip/hip_runtime.h>
#include <hip/hip_bf16.h>
#include <math.h>

typedef __attribute__((ext_vector_type(16))) _Float16 v16h;
typedef __attribute__((ext_vector_type(8)))  _Float16 v8h;
typedef __attribute__((ext_vector_type(4)))  _Float16 v4h;
typedef __attribute__((ext_vector_type(8)))  float    v8f;

#define B_ROWS 8192
#define HDIM   1024
#define NDIM   4096   // 4*H
#define KDIM   1024

// GEMM tiling: block 128(M) x 256(N), BK=64, 8 waves in 2x4, wave tile 64x64
#define BM     128
#define BN     256
#define BK     64
#define LDSTR  72                 // halfs per LDS row (144 B -> 36-bank spread)
#define A_HALF (BM * LDSTR)       // 9216 halfs per A buffer
#define B_HALF (BN * LDSTR)       // 18432 halfs per B buffer
#define B_OFF  (2 * A_HALF)       // B buffers start after both A buffers
#define SMEM_HALFS (2 * A_HALF + 2 * B_HALF)   // 55296 halfs = 108 KB

// ---------------------------------------------------------------------------
// fp32 -> f16 conversion (one pass; inputs stay L2-resident afterwards)
// ---------------------------------------------------------------------------
__global__ __launch_bounds__(256)
void cvt_f32_f16(const float* __restrict__ s, _Float16* __restrict__ d, int n4) {
    int i = blockIdx.x * 256 + threadIdx.x;
    if (i < n4) {
        float4 v = ((const float4*)s)[i];
        v4h h;
        h[0] = (_Float16)v.x; h[1] = (_Float16)v.y;
        h[2] = (_Float16)v.z; h[3] = (_Float16)v.w;
        ((v4h*)d)[i] = h;
    }
}

// ---------------------------------------------------------------------------
// Fragment load from LDS, CDNA5 16-bit A-matrix 16x32 layout:
// lane L<16: M=L, K={0..7,16..23}; lane L>=16: M=L-16, K={8..15,24..31}.
// ---------------------------------------------------------------------------
static __device__ inline v16h load_frag16x32(const _Float16* rowBase, int khalf) {
    v8h lo = *(const v8h*)(rowBase + khalf);
    v8h hi = *(const v8h*)(rowBase + 16 + khalf);
    v16h r;
#pragma unroll
    for (int i = 0; i < 8; ++i) { r[i] = lo[i]; r[i + 8] = hi[i]; }
    return r;
}

static __device__ inline void async_b128(unsigned ldsAddr, const _Float16* g) {
    asm volatile("global_load_async_to_lds_b128 %0, %1, off"
                 :: "v"(ldsAddr),
                    "v"((unsigned long long)(uintptr_t)g)
                 : "memory");
}

// ---------------------------------------------------------------------------
// GEMM: C[M,N] = A[M,K] * W[N,K]^T, f16 operands, f32 WMMA accumulate.
// Double-buffered LDS filled by async-to-LDS DMA (ASYNCcnt), 2x4 wave grid.
// ---------------------------------------------------------------------------
__global__ __launch_bounds__(256)
void wmma_gemm_async(const _Float16* __restrict__ A,
                     const _Float16* __restrict__ W,
                     float* __restrict__ C) {
    __shared__ __align__(16) _Float16 smem[SMEM_HALFS];

    const int tid  = threadIdx.x;
    const int lane = tid & 31;
    const int wid  = tid >> 5;      // 0..7
    const int wm   = wid & 1;       // 2 row-groups of 64
    const int wn   = wid >> 1;      // 4 col-groups of 64

    const int blockN = blockIdx.x * BN;
    const int blockM = blockIdx.y * BM;

    const unsigned ldsBase = (unsigned)(uintptr_t)(void*)smem;

    // 12 async b128 per thread per K-tile: 4 for A (128x64 f16), 8 for B (256x64)
    auto issue_tile = [&](int buf, int ko) {
#pragma unroll
        for (int i = 0; i < 4; ++i) {
            int chunk = tid + 256 * i;            // 1024 chunks of 16B
            int row = chunk >> 3;
            int c   = (chunk & 7) * 8;            // half offset in row
            const _Float16* g = A + (size_t)(blockM + row) * KDIM + ko * BK + c;
            unsigned l = ldsBase + (unsigned)((buf * A_HALF + row * LDSTR + c) * 2);
            async_b128(l, g);
        }
#pragma unroll
        for (int i = 0; i < 8; ++i) {
            int chunk = tid + 256 * i;            // 2048 chunks of 16B
            int row = chunk >> 3;
            int c   = (chunk & 7) * 8;
            const _Float16* g = W + (size_t)(blockN + row) * KDIM + ko * BK + c;
            unsigned l = ldsBase + (unsigned)((B_OFF + buf * B_HALF + row * LDSTR + c) * 2);
            async_b128(l, g);
        }
    };

    v8f acc[4][4];
#pragma unroll
    for (int mi = 0; mi < 4; ++mi)
#pragma unroll
        for (int nj = 0; nj < 4; ++nj) acc[mi][nj] = {};

    const int lane15 = lane & 15;
    const int khalf  = (lane >> 4) * 8;
    const int NK     = KDIM / BK;   // 16

    issue_tile(0, 0);

    for (int ko = 0; ko < NK; ++ko) {
        const int buf = ko & 1;
        if (ko + 1 < NK) {
            issue_tile(buf ^ 1, ko + 1);          // overlap next tile DMA with compute
            asm volatile("s_wait_asynccnt 0xc" ::: "memory");  // current tile done
        } else {
            asm volatile("s_wait_asynccnt 0x0" ::: "memory");
        }
        __syncthreads();

        const _Float16* sA = smem + buf * A_HALF;
        const _Float16* sB = smem + B_OFF + buf * B_HALF;

#pragma unroll
        for (int ks = 0; ks < 2; ++ks) {          // two 16x16x32 steps per BK=64
            v16h bfr[4];
#pragma unroll
            for (int nj = 0; nj < 4; ++nj)
                bfr[nj] = load_frag16x32(sB + (wn * 64 + nj * 16 + lane15) * LDSTR + ks * 32, khalf);
#pragma unroll
            for (int mi = 0; mi < 4; ++mi) {
                v16h af = load_frag16x32(sA + (wm * 64 + mi * 16 + lane15) * LDSTR + ks * 32, khalf);
#pragma unroll
                for (int nj = 0; nj < 4; ++nj)
                    acc[mi][nj] = __builtin_amdgcn_wmma_f32_16x16x32_f16(
                        false, af, false, bfr[nj],
                        (short)0, acc[mi][nj], false, false);
            }
        }
        __syncthreads();
    }

    // store: lanes 0-15 -> M=r,N=lane; lanes 16-31 -> M=8+r,N=lane-16
    const int mOff = (lane >> 4) * 8;
#pragma unroll
    for (int mi = 0; mi < 4; ++mi) {
#pragma unroll
        for (int nj = 0; nj < 4; ++nj) {
            const int n = blockN + wn * 64 + nj * 16 + lane15;
            const int mBase = blockM + wm * 64 + mi * 16 + mOff;
#pragma unroll
            for (int r = 0; r < 8; ++r)
                C[(size_t)(mBase + r) * NDIM + n] = acc[mi][nj][r];
        }
    }
}

// ---------------------------------------------------------------------------
// Fused LN + gates kernel: one block per batch row (256 threads, 4 cols each).
// ---------------------------------------------------------------------------
static __device__ inline void block_reduce2(float& s, float& q) {
    __shared__ float rs[8], rq[8];
    const int lane = threadIdx.x & 31;
    const int wid  = threadIdx.x >> 5;
#pragma unroll
    for (int off = 16; off > 0; off >>= 1) {
        s += __shfl_down(s, off, 32);
        q += __shfl_down(q, off, 32);
    }
    if (lane == 0) { rs[wid] = s; rq[wid] = q; }
    __syncthreads();
    float ts = 0.f, tq = 0.f;
#pragma unroll
    for (int w = 0; w < 8; ++w) { ts += rs[w]; tq += rq[w]; }
    __syncthreads();
    s = ts; q = tq;
}

static __device__ inline float sigmoidf_(float x) { return 1.0f / (1.0f + expf(-x)); }

__global__ __launch_bounds__(256)
void ln_lstm_gates(const float* __restrict__ pre_ih,
                   const float* __restrict__ pre_hh,
                   const float* __restrict__ c_in,
                   const float* __restrict__ b_ih,
                   const float* __restrict__ ln_gamma,
                   const float* __restrict__ ln_beta,
                   float* __restrict__ out_h,
                   float* __restrict__ out_c) {
    const int row  = blockIdx.x;
    const int tid  = threadIdx.x;
    const int col0 = tid * 4;
    const float invH = 1.0f / (float)HDIM;
    const float eps  = 1e-5f;

    float4 ih[4], hh[4];
    float muI[4], rsI[4], muH[4], rsH[4];
#pragma unroll
    for (int k = 0; k < 4; ++k) {
        ih[k] = *(const float4*)(pre_ih + (size_t)row * NDIM + k * HDIM + col0);
        hh[k] = *(const float4*)(pre_hh + (size_t)row * NDIM + k * HDIM + col0);
    }
#pragma unroll
    for (int k = 0; k < 4; ++k) {
        float s = ih[k].x + ih[k].y + ih[k].z + ih[k].w;
        float q = ih[k].x*ih[k].x + ih[k].y*ih[k].y + ih[k].z*ih[k].z + ih[k].w*ih[k].w;
        block_reduce2(s, q);
        muI[k] = s * invH;
        rsI[k] = rsqrtf(fmaxf(q * invH - muI[k] * muI[k], 0.0f) + eps);

        s = hh[k].x + hh[k].y + hh[k].z + hh[k].w;
        q = hh[k].x*hh[k].x + hh[k].y*hh[k].y + hh[k].z*hh[k].z + hh[k].w*hh[k].w;
        block_reduce2(s, q);
        muH[k] = s * invH;
        rsH[k] = rsqrtf(fmaxf(q * invH - muH[k] * muH[k], 0.0f) + eps);
    }

    float4 cv = *(const float4*)(c_in + (size_t)row * HDIM + col0);
    float cold[4] = {cv.x, cv.y, cv.z, cv.w};
    float ihv[4][4] = {{ih[0].x,ih[0].y,ih[0].z,ih[0].w},{ih[1].x,ih[1].y,ih[1].z,ih[1].w},
                       {ih[2].x,ih[2].y,ih[2].z,ih[2].w},{ih[3].x,ih[3].y,ih[3].z,ih[3].w}};
    float hhv[4][4] = {{hh[0].x,hh[0].y,hh[0].z,hh[0].w},{hh[1].x,hh[1].y,hh[1].z,hh[1].w},
                       {hh[2].x,hh[2].y,hh[2].z,hh[2].w},{hh[3].x,hh[3].y,hh[3].z,hh[3].w}};

    float gate[4][4];   // i, f, a, o
#pragma unroll
    for (int k = 0; k < 4; ++k) {
#pragma unroll
        for (int j = 0; j < 4; ++j) {
            const int col = col0 + j;
            const float gI = ln_gamma[(2*k+0) * HDIM + col];
            const float bI = ln_beta [(2*k+0) * HDIM + col];
            const float gH = ln_gamma[(2*k+1) * HDIM + col];
            const float bH = ln_beta [(2*k+1) * HDIM + col];
            const float bias = b_ih[k * HDIM + col];
            const float z = (ihv[k][j] - muI[k]) * rsI[k] * gI + bI
                          + (hhv[k][j] - muH[k]) * rsH[k] * gH + bH + bias;
            gate[k][j] = (k == 2) ? tanhf(z) : sigmoidf_(z);
        }
    }

    float cnew[4];
    float s = 0.f, q = 0.f;
#pragma unroll
    for (int j = 0; j < 4; ++j) {
        cnew[j] = gate[1][j] * cold[j] + gate[0][j] * gate[2][j];
        s += cnew[j];
        q += cnew[j] * cnew[j];
    }
    block_reduce2(s, q);
    const float muC = s * invH;
    const float rsC = rsqrtf(fmaxf(q * invH - muC * muC, 0.0f) + eps);

    float4 ho, co;
    float hv[4];
#pragma unroll
    for (int j = 0; j < 4; ++j) {
        const int col = col0 + j;
        const float gC = ln_gamma[8 * HDIM + col];
        const float bC = ln_beta [8 * HDIM + col];
        const float lc = (cnew[j] - muC) * rsC * gC + bC;
        hv[j] = gate[3][j] * tanhf(lc);
    }
    ho.x = hv[0]; ho.y = hv[1]; ho.z = hv[2]; ho.w = hv[3];
    co.x = cnew[0]; co.y = cnew[1]; co.z = cnew[2]; co.w = cnew[3];
    *(float4*)(out_h + (size_t)row * HDIM + col0) = ho;
    *(float4*)(out_c + (size_t)row * HDIM + col0) = co;
}

// ---------------------------------------------------------------------------
extern "C" void kernel_launch(void* const* d_in, const int* in_sizes, int n_in,
                              void* d_out, int out_size, void* d_ws, size_t ws_size,
                              hipStream_t stream) {
    const float* x        = (const float*)d_in[0];   // [8192,1024]
    const float* h        = (const float*)d_in[1];   // [8192,1024]
    const float* c        = (const float*)d_in[2];   // [8192,1024]
    const float* w_ih     = (const float*)d_in[3];   // [4096,1024]
    const float* w_hh     = (const float*)d_in[4];   // [4096,1024]
    const float* b_ih     = (const float*)d_in[5];   // [4096]
    const float* ln_gamma = (const float*)d_in[6];   // [9,1024]
    const float* ln_beta  = (const float*)d_in[7];   // [9,1024]

    char* ws = (char*)d_ws;
    const size_t preBytes = (size_t)B_ROWS * NDIM * sizeof(float);      // 128 MB
    const size_t xBytes16 = (size_t)B_ROWS * KDIM * sizeof(_Float16);   // 16 MB
    const size_t wBytes16 = (size_t)NDIM * KDIM * sizeof(_Float16);     //  8 MB

    float*     pre_ih = (float*)(ws);
    float*     pre_hh = (float*)(ws + preBytes);
    _Float16*  x16    = (_Float16*)(ws + 2 * preBytes);
    _Float16*  h16    = (_Float16*)(ws + 2 * preBytes + xBytes16);
    _Float16*  wih16  = (_Float16*)(ws + 2 * preBytes + 2 * xBytes16);
    _Float16*  whh16  = (_Float16*)(ws + 2 * preBytes + 2 * xBytes16 + wBytes16);

    float* out   = (float*)d_out;
    float* out_h = out;
    float* out_c = out + (size_t)B_ROWS * HDIM;

    const int nx4 = B_ROWS * KDIM / 4;   // 2,097,152
    const int nw4 = NDIM * KDIM / 4;     // 1,048,576
    cvt_f32_f16<<<(nx4 + 255) / 256, 256, 0, stream>>>(x, x16, nx4);
    cvt_f32_f16<<<(nx4 + 255) / 256, 256, 0, stream>>>(h, h16, nx4);
    cvt_f32_f16<<<(nw4 + 255) / 256, 256, 0, stream>>>(w_ih, wih16, nw4);
    cvt_f32_f16<<<(nw4 + 255) / 256, 256, 0, stream>>>(w_hh, whh16, nw4);

    dim3 gemmGrid(NDIM / BN, B_ROWS / BM);    // 16 x 64
    wmma_gemm_async<<<gemmGrid, 256, 0, stream>>>(x16, wih16, pre_ih);
    wmma_gemm_async<<<gemmGrid, 256, 0, stream>>>(h16, whh16, pre_hh);

    ln_lstm_gates<<<B_ROWS, 256, 0, stream>>>(pre_ih, pre_hh, c, b_ih,
                                              ln_gamma, ln_beta, out_h, out_c);
}
